// LSTMAL_55843164782739
// MI455X (gfx1250) — matrix operand order repaired and checked by reading, not compiled, not run on hardware
//
#include <hip/hip_runtime.h>
#include <hip/hip_bf16.h>
#include <stdint.h>

// Problem constants (fixed by the reference)
#define SLEN  256
#define BATCH 64
#define DIM   512
#define HID   512
#define G4    2048   // 4*HID

typedef __bf16 bf16_t;
typedef __attribute__((ext_vector_type(16))) __bf16    v16bf;
typedef __attribute__((ext_vector_type(8)))  float     v8f;
typedef __attribute__((ext_vector_type(4)))  uint32_t  u32x4;   // builtin vector: AS-safe

// explicit global-address-space pointers -> global_load/global_store (not flat_*)
typedef __attribute__((address_space(1))) const u32x4*  g_cu4p;
typedef __attribute__((address_space(1))) const float*  g_cfp;
typedef __attribute__((address_space(1))) float*        g_fp;
typedef __attribute__((address_space(1))) const bf16_t* g_cbfp;
typedef __attribute__((address_space(1))) bf16_t*       g_bfp;
typedef __attribute__((address_space(1))) const int*    g_cip;

struct U32x8 { u32x4 lo, hi; };   // 32 bytes == one v16bf fragment

__device__ __forceinline__ bf16_t f2bf(float f) {
  uint32_t u = __builtin_bit_cast(uint32_t, f);
  uint32_t r = (u + 0x7FFFu + ((u >> 16) & 1u)) >> 16;     // RNE
  unsigned short s = (unsigned short)r;
  return __builtin_bit_cast(bf16_t, s);
}
__device__ __forceinline__ float bf2f(bf16_t b) {
  unsigned short s = __builtin_bit_cast(unsigned short, b);
  uint32_t u = ((uint32_t)s) << 16;
  return __builtin_bit_cast(float, u);
}
__device__ __forceinline__ float sigm(float x) { return 1.0f / (1.0f + expf(-x)); }

// ---- WMMA fragment loaders (documented CDNA5 wave32 layouts) ----
// A (16x32 bf16, MxK): lane l: m=l&15, half=l>>4; 8 contiguous K at half*8, 8 more at +16.
__device__ __forceinline__ v16bf load_a_g(g_cbfp p) {
  U32x8 u;
  u.lo = *(g_cu4p)(p);
  u.hi = *(g_cu4p)(p + 16);
  return __builtin_bit_cast(v16bf, u);
}
// B pre-packed fragment-major: frag = (ntile*(K/32)+kchunk), 32 lanes x 32B contiguous.
__device__ __forceinline__ v16bf load_b_g(g_cbfp p) {
  U32x8 u;
  u.lo = *(g_cu4p)(p);
  u.hi = *(g_cu4p)(p + 8);
  return __builtin_bit_cast(v16bf, u);
}
__device__ __forceinline__ v8f wmma_bf(v16bf a, v16bf b, v8f c) {
  return __builtin_amdgcn_wmma_f32_16x16x32_bf16(false, a, false, b, (short)0, c, false, false);
}

// ---------------- shared LSTM cell body ----------------
// gates(64x2048) = [seg0|seg1(|seg2)](64 x nseg*512) @ Bp ; then i,f,g,o nonlinearity.
// blockIdx.x = per-gate 16-col group (0..31); block = 256 threads (8 waves, 2 M-tiles each).
// GEMM loop is software-pipelined: next chunk's fragments load while current WMMAs issue.
__device__ __forceinline__ void cell_body(
    const bf16_t* s0, int l0, const bf16_t* s1, int l1,
    const bf16_t* s2, int l2, int nseg,
    const bf16_t* Bp, const float* bias,
    float* c_state, float* h_out, bf16_t* hbf_out, bf16_t* oy_bf,
    float* lds_g) {
  const int tid = threadIdx.x, lane = tid & 31, w = tid >> 5;
  const int j = blockIdx.x;            // gate-column group 0..31
  const int g = w >> 1;                // gate 0..3
  const int mt0 = (w & 1) * 2;         // M-tile pair
  const int ntile = g * 32 + j;        // of 128 N-tiles
  const int kh = lane >> 4;
  const int arow0 = mt0 * 16 + (lane & 15);
  const int arow1 = arow0 + 16;
  const int nch = nseg * 16;
  const bf16_t* segp[3] = { s0, s1, s2 };
  const int     segld[3] = { l0, l1, l2 };

  g_cbfp bw = (g_cbfp)Bp + ((size_t)ntile * nch) * 512 + lane * 16;
  v8f acc0 = {}; v8f acc1 = {};
  for (int sg = 0; sg < nseg; ++sg) {
    g_cbfp pa0 = (g_cbfp)segp[sg] + (size_t)arow0 * segld[sg] + kh * 8;
    g_cbfp pa1 = (g_cbfp)segp[sg] + (size_t)arow1 * segld[sg] + kh * 8;
    v16bf a0 = load_a_g(pa0);
    v16bf a1 = load_a_g(pa1);
    v16bf bb = load_b_g(bw);
#pragma unroll 5
    for (int kc = 0; kc < 15; ++kc) {
      v16bf na0 = load_a_g(pa0 + (kc + 1) * 32);
      v16bf na1 = load_a_g(pa1 + (kc + 1) * 32);
      v16bf nbb = load_b_g(bw + 512);
      acc0 = wmma_bf(a0, bb, acc0);
      acc1 = wmma_bf(a1, bb, acc1);
      a0 = na0; a1 = na1; bb = nbb;
      bw += 512;
    }
    acc0 = wmma_bf(a0, bb, acc0);
    acc1 = wmma_bf(a1, bb, acc1);
    bw += 512;
  }
  {
    int col = lane & 15, rb = (lane >> 4) * 8;
    float* gp = &lds_g[g * 1024];
#pragma unroll
    for (int r = 0; r < 8; ++r) {
      gp[(mt0 * 16 + rb + r) * 16 + col]       = acc0[r];
      gp[((mt0 + 1) * 16 + rb + r) * 16 + col] = acc1[r];
    }
  }
  __syncthreads();
  g_cfp  bg  = (g_cfp)bias;
  g_fp   cg  = (g_fp)c_state;
  g_fp   hg  = (g_fp)h_out;
  g_bfp  hbg = (g_bfp)hbf_out;
  g_bfp  oyg = (g_bfp)oy_bf;
#pragma unroll
  for (int i = 0; i < 4; ++i) {
    int e = tid + i * 256;             // 64 rows x 16 cols
    int row = e >> 4, col = e & 15;
    int n = j * 16 + col;              // within-gate column 0..511
    float gi = lds_g[0 * 1024 + row * 16 + col] + bg[n];
    float gf = lds_g[1 * 1024 + row * 16 + col] + bg[512 + n];
    float gg = lds_g[2 * 1024 + row * 16 + col] + bg[1024 + n];
    float go = lds_g[3 * 1024 + row * 16 + col] + bg[1536 + n];
    float cp = cg[row * 512 + n];
    float cn = sigm(gf) * cp + sigm(gi) * tanhf(gg);
    float hn = sigm(go) * tanhf(cn);
    cg[row * 512 + n]  = cn;
    hg[row * 512 + n]  = hn;
    hbg[row * 512 + n] = f2bf(hn);
    if (oy_bf) oyg[(size_t)row * 1024 + n] = f2bf(hn);
  }
}

// 4 independent encoder chains in one launch: grid = (32, 4)
struct Enc4Args {
  const bf16_t* xseg[4];
  const bf16_t* hin[4];
  bf16_t*       hout[4];
  float*        c[4];
  float*        h[4];
  bf16_t*       oy[4];
  const bf16_t* Bp[4];
  const float*  bias[4];
};
__global__ __launch_bounds__(256) void enc4_kernel(Enc4Args a) {
  __shared__ float lds_g[4 * 64 * 16];
  int ch = blockIdx.y;
  cell_body(a.xseg[ch], 512, a.hin[ch], 512, nullptr, 0, 2,
            a.Bp[ch], a.bias[ch], a.c[ch], a.h[ch], a.hout[ch], a.oy[ch], lds_g);
}

// decoder cell: grid = 32
__global__ __launch_bounds__(256) void cell_kernel(
    const bf16_t* s0, const bf16_t* s1, const bf16_t* s2,
    const bf16_t* Bp, const float* bias,
    float* c_state, float* h_out, bf16_t* hbf_out) {
  __shared__ float lds_g[4 * 64 * 16];
  cell_body(s0, 512, s1, 512, s2, 512, 3, Bp, bias,
            c_state, h_out, hbf_out, nullptr, lds_g);
}

// ---------------- generic 64xK @ Kx512 GEMM (+bias, +tanh, +fused sq-loss) ----
// grid=16, block=256 (8 waves: 2 N-tiles x 4 M-tiles).
__global__ __launch_bounds__(256) void gemm64_kernel(
    const bf16_t* s0, int l0, const bf16_t* s1, int l1,
    const bf16_t* s2, int l2, int nseg,
    const bf16_t* Bp, const float* bias,
    float* out_pre, float* out_act, bf16_t* out_act_bf, int do_tanh,
    const float* lref, float* lpart) {
  __shared__ float red[256];
  const int tid = threadIdx.x, lane = tid & 31, w = tid >> 5;
  const int mt = w & 3;
  const int ntile = blockIdx.x * 2 + (w >> 2);
  const int kh = lane >> 4;
  const int arow = mt * 16 + (lane & 15);
  const int nch = nseg * 16;
  const bf16_t* segp[3] = { s0, s1, s2 };
  const int     segld[3] = { l0, l1, l2 };
  g_cbfp bw = (g_cbfp)Bp + ((size_t)ntile * nch) * 512 + lane * 16;
  v8f acc = {};
  for (int sg = 0; sg < nseg; ++sg) {
    g_cbfp pa = (g_cbfp)segp[sg] + (size_t)arow * segld[sg] + kh * 8;
    v16bf a  = load_a_g(pa);
    v16bf bb = load_b_g(bw);
#pragma unroll 5
    for (int kc = 0; kc < 15; ++kc) {
      v16bf na  = load_a_g(pa + (kc + 1) * 32);
      v16bf nbb = load_b_g(bw + 512);
      acc = wmma_bf(a, bb, acc);
      a = na; bb = nbb;
      bw += 512;
    }
    acc = wmma_bf(a, bb, acc);
    bw += 512;
  }
  int col = ntile * 16 + (lane & 15);
  int rb = mt * 16 + (lane >> 4) * 8;
  g_cfp bg = (g_cfp)bias;
  float bv = bias ? bg[col] : 0.0f;
  float ls = 0.0f;
  g_fp  pre = (g_fp)out_pre;
  g_fp  act = (g_fp)out_act;
  g_bfp abf = (g_bfp)out_act_bf;
  g_cfp lrf = (g_cfp)lref;
#pragma unroll
  for (int r = 0; r < 8; ++r) {
    int row = rb + r;
    float v = acc[r] + bv;
    if (out_pre) pre[row * 512 + col] = v;
    float a = do_tanh ? tanhf(v) : v;
    if (out_act)    act[row * 512 + col] = a;
    if (out_act_bf) abf[row * 512 + col] = f2bf(a);
    if (lref) { float d = a - lrf[row * 512 + col]; ls += d * d; }
  }
  if (lpart) {
    red[tid] = ls; __syncthreads();
    for (int o = 128; o > 0; o >>= 1) { if (tid < o) red[tid] += red[tid + o]; __syncthreads(); }
    if (tid == 0) ((g_fp)lpart)[blockIdx.x] = red[0];
  }
}

// src_att[b,s,h] = sum_e out_y[s,b,e]*att_src_W[h,e]  -- big (16384x1024)@(1024x512)
// grid=(32,128), 8 waves/block, one M-tile per wave.
__global__ __launch_bounds__(256) void srcatt_kernel(const bf16_t* A, const bf16_t* Bp,
                                                     float* out) {
  const int tid = threadIdx.x, lane = tid & 31, w = tid >> 5;
  const int mtile = blockIdx.y * 8 + w;     // 0..1023
  const int ntile = blockIdx.x;             // 0..31
  const int kh = lane >> 4;
  const int arow = mtile * 16 + (lane & 15);
  g_cbfp pa = (g_cbfp)A + (size_t)arow * 1024 + kh * 8;
  g_cbfp bw = (g_cbfp)Bp + ((size_t)ntile * 32) * 512 + lane * 16;
  v8f acc = {};
  v16bf a  = load_a_g(pa);
  v16bf bb = load_b_g(bw);
#pragma unroll 8
  for (int kc = 0; kc < 31; ++kc) {         // K=1024, pipelined
    v16bf na  = load_a_g(pa + (kc + 1) * 32);
    v16bf nbb = load_b_g(bw + 512);
    acc = wmma_bf(a, bb, acc);
    a = na; bb = nbb;
    bw += 512;
  }
  acc = wmma_bf(a, bb, acc);
  int col = ntile * 16 + (lane & 15);
  int m0 = mtile * 16 + (lane >> 4) * 8;
  g_fp og = (g_fp)out;
#pragma unroll
  for (int r = 0; r < 8; ++r) {
    int m = m0 + r, b = m & 63, s = m >> 6;
    og[((size_t)b * SLEN + s) * HID + col] = acc[r];
  }
}

// scores -> softmax -> ctx (bf16 for the att_vec GEMM). grid=64 (one per batch).
__global__ __launch_bounds__(256) void attn_kernel(const float* src_att, const float* h,
                                                   const bf16_t* out_y_bf, const int* lens,
                                                   bf16_t* ctx_bf) {
  __shared__ float hs[512];
  __shared__ float alpha[256];
  __shared__ float red[256];
  int b = blockIdx.x, tid = threadIdx.x;
  g_cfp hgp = (g_cfp)h;
  hs[tid]       = hgp[b * 512 + tid];
  hs[tid + 256] = hgp[b * 512 + tid + 256];
  __syncthreads();
  g_cfp sa = (g_cfp)src_att + ((size_t)b * SLEN + tid) * HID;
  float sc = 0.f;
  for (int k = 0; k < 512; ++k) sc += sa[k] * hs[k];
  if (tid >= ((g_cip)lens)[b]) sc = -1e9f;
  red[tid] = sc; __syncthreads();
  for (int o = 128; o > 0; o >>= 1) { if (tid < o) red[tid] = fmaxf(red[tid], red[tid + o]); __syncthreads(); }
  float mx = red[0]; __syncthreads();
  float ex = expf(sc - mx);
  red[tid] = ex; __syncthreads();
  for (int o = 128; o > 0; o >>= 1) { if (tid < o) red[tid] += red[tid + o]; __syncthreads(); }
  alpha[tid] = ex / red[0];
  __syncthreads();
  g_cbfp oy = (g_cbfp)out_y_bf;
  g_bfp  cb = (g_bfp)ctx_bf;
  for (int i = 0; i < 4; ++i) {
    int e = tid + i * 256;                  // 0..1023
    float acc = 0.f;
    for (int s = 0; s < SLEN; ++s)
      acc += alpha[s] * bf2f(oy[((size_t)s * BATCH + b) * 1024 + e]);
    cb[b * 1024 + e] = f2bf(acc);
  }
}

// per-block partial of sum((a-b)^2) over 32768 elems; grid=32.
__global__ __launch_bounds__(256) void losssq_kernel(const float* a, const float* bp,
                                                     float* part) {
  __shared__ float red[256];
  int tid = threadIdx.x;
  g_cfp ag = (g_cfp)a, bg = (g_cfp)bp;
  float acc = 0.f;
  for (int i = 0; i < 4; ++i) {
    int e = blockIdx.x * 1024 + tid + i * 256;
    float d = ag[e] - bg[e];
    acc += d * d;
  }
  red[tid] = acc; __syncthreads();
  for (int o = 128; o > 0; o >>= 1) { if (tid < o) red[tid] += red[tid + o]; __syncthreads(); }
  if (tid == 0) ((g_fp)part)[blockIdx.x] = red[0];
}

__global__ void final_kernel(const float* lb, const float* ld, float* out) {
  __shared__ float red[256];
  int tid = threadIdx.x;
  g_cfp lbg = (g_cfp)lb, ldg = (g_cfp)ld;
  float a = 0.f;
  for (int i = tid; i < 64; i += 256) a += lbg[i];
  float d = 0.f;
  for (int i = tid; i < SLEN * 16; i += 256) d += ldg[i];
  red[tid] = a / (2.0f * 64.0f * 512.0f) + d / ((float)SLEN * 64.0f * 512.0f);
  __syncthreads();
  for (int o = 128; o > 0; o >>= 1) { if (tid < o) red[tid] += red[tid + o]; __syncthreads(); }
  if (tid == 0) out[0] = red[0];
}

// ---------------- utility kernels ----------------
__global__ void zero_kernel(uint32_t* p, int n) {
  int i = blockIdx.x * blockDim.x + threadIdx.x;
  if (i < n) ((g_fp)(float*)p)[i] = 0.0f;
}
__global__ void f2bf_kernel(const float* in, bf16_t* out, int n) {
  int i = blockIdx.x * blockDim.x + threadIdx.x;
  if (i < n) ((g_bfp)out)[i] = f2bf(((g_cfp)in)[i]);
}
__global__ void emb_kernel(const float* table, const int* tgt, bf16_t* out) {
  int i = blockIdx.x * blockDim.x + threadIdx.x;
  if (i >= SLEN * BATCH * DIM) return;
  int tb = i >> 9, col = i & 511;
  int tok = ((g_cip)tgt)[tb];
  ((g_bfp)out)[i] = f2bf(((g_cfp)table)[(size_t)tok * DIM + col]);
}
// Pack W (given as weight[n][k], gemm needs B[k][n]) into WMMA B-fragment layout.
__global__ void pack_w_kernel(const float* w0, int ld0, const float* w1, int ld1,
                              int split, int K, bf16_t* out, int total) {
  int idx = blockIdx.x * blockDim.x + threadIdx.x;
  if (idx >= total) return;
  int le   = idx & 15;
  int lane = (idx >> 4) & 31;
  int frag = idx >> 9;
  int nch  = K >> 5;
  int ntile  = frag / nch;
  int kchunk = frag - ntile * nch;
  int k = kchunk * 32 + (lane >> 4) * 8 + (le & 7) + (le >> 3) * 16;
  int n = ntile * 16 + (lane & 15);
  float v = (k < split) ? ((g_cfp)w0)[(size_t)n * ld0 + k]
                        : ((g_cfp)w1)[(size_t)n * ld1 + (k - split)];
  ((g_bfp)out)[idx] = f2bf(v);
}

// ---------------- host driver ----------------
extern "C" void kernel_launch(void* const* d_in, const int* in_sizes, int n_in,
                              void* d_out, int out_size, void* d_ws, size_t ws_size,
                              hipStream_t stream) {
  (void)in_sizes; (void)n_in; (void)out_size; (void)ws_size;
  const float* x        = (const float*)d_in[0];
  const float* y        = (const float*)d_in[1];
  const float* ef_Wih   = (const float*)d_in[2];
  const float* ef_Whh   = (const float*)d_in[3];
  const float* ef_b     = (const float*)d_in[4];
  const float* eg_Wih   = (const float*)d_in[5];
  const float* eg_Whh   = (const float*)d_in[6];
  const float* eg_b     = (const float*)d_in[7];
  const float* gi_W     = (const float*)d_in[8];
  const float* gi_b     = (const float*)d_in[9];
  const float* dec_Wih  = (const float*)d_in[10];
  const float* dec_Whh  = (const float*)d_in[11];
  const float* dec_b    = (const float*)d_in[12];
  const float* asrc_W   = (const float*)d_in[13];
  const float* avec_W   = (const float*)d_in[14];
  const float* emb_tab  = (const float*)d_in[15];
  const int*   tgt      = (const int*)d_in[16];
  const int*   lens     = (const int*)d_in[17];
  float* out = (float*)d_out;

  // workspace carve-up (~160 MB)
  char* W = (char*)d_ws;
  size_t off = 0;
  auto ALLOC = [&](size_t bytes) { size_t r = off; off += (bytes + 255) & ~(size_t)255; return r; };
  bf16_t* Bp_enc[4];
  for (int i = 0; i < 4; ++i) Bp_enc[i] = (bf16_t*)(W + ALLOC((size_t)1024 * G4 * 2));
  bf16_t* Bp_dec  = (bf16_t*)(W + ALLOC((size_t)1536 * G4 * 2));
  bf16_t* Bp_gini = (bf16_t*)(W + ALLOC((size_t)1024 * 512 * 2));
  bf16_t* Bp_asrc = (bf16_t*)(W + ALLOC((size_t)1024 * 512 * 2));
  bf16_t* Bp_avec = (bf16_t*)(W + ALLOC((size_t)1536 * 512 * 2));
  bf16_t* x_bf    = (bf16_t*)(W + ALLOC((size_t)SLEN * BATCH * DIM * 2));
  bf16_t* y_bf    = (bf16_t*)(W + ALLOC((size_t)SLEN * BATCH * DIM * 2));
  bf16_t* emb_bf  = (bf16_t*)(W + ALLOC((size_t)SLEN * BATCH * DIM * 2));
  bf16_t* outy_bf = (bf16_t*)(W + ALLOC((size_t)SLEN * BATCH * 1024 * 2));
  float*  src_att = (float*)(W + ALLOC((size_t)BATCH * SLEN * HID * 4));
  float*  h_st[4]; bf16_t* hbf[4][2]; float* c_st[4];
  for (int i = 0; i < 4; ++i) {
    h_st[i]   = (float*)(W + ALLOC(64 * 512 * 4));
    hbf[i][0] = (bf16_t*)(W + ALLOC(64 * 512 * 2));
    hbf[i][1] = (bf16_t*)(W + ALLOC(64 * 512 * 2));
    c_st[i]   = (float*)(W + ALLOC(64 * 512 * 4));
  }
  float*  dec_h   = (float*)(W + ALLOC(64 * 512 * 4));
  bf16_t* dhbf[2] = { (bf16_t*)(W + ALLOC(64 * 512 * 2)), (bf16_t*)(W + ALLOC(64 * 512 * 2)) };
  float*  dec_c   = (float*)(W + ALLOC(64 * 512 * 4));
  bf16_t* cyf_bf  = (bf16_t*)(W + ALLOC(64 * 512 * 2));
  bf16_t* cyb_bf  = (bf16_t*)(W + ALLOC(64 * 512 * 2));
  float*  att_f   = (float*)(W + ALLOC(64 * 512 * 4));
  bf16_t* att_bf  = (bf16_t*)(W + ALLOC(64 * 512 * 2));
  bf16_t* ctx_bf  = (bf16_t*)(W + ALLOC(64 * 1024 * 2));
  float*  lb      = (float*)(W + ALLOC(64 * 4));
  float*  ldp     = (float*)(W + ALLOC((size_t)SLEN * 16 * 4));

  auto NB = [](int n) { return (n + 255) / 256; };

  // 0) zero initial states (ws is poisoned; must re-zero every call)
  for (int i = 0; i < 4; ++i) {
    zero_kernel<<<NB(16384), 256, 0, stream>>>((uint32_t*)hbf[i][0], 16384);
    zero_kernel<<<NB(32768), 256, 0, stream>>>((uint32_t*)c_st[i], 32768);
  }
  zero_kernel<<<NB(16384), 256, 0, stream>>>((uint32_t*)att_bf, 16384);

  // 1) activations -> bf16, embedding gather
  const int NXY = SLEN * BATCH * DIM;
  f2bf_kernel<<<NB(NXY), 256, 0, stream>>>(x, x_bf, NXY);
  f2bf_kernel<<<NB(NXY), 256, 0, stream>>>(y, y_bf, NXY);
  emb_kernel<<<NB(NXY), 256, 0, stream>>>(emb_tab, tgt, emb_bf);

  // 2) pack weights into WMMA B-fragment layout (B[k][n] = weight[n][k])
  const float* Wih_s[4] = { ef_Wih, ef_Wih + (size_t)G4 * DIM, eg_Wih, eg_Wih + (size_t)G4 * DIM };
  const float* Whh_s[4] = { ef_Whh, ef_Whh + (size_t)G4 * HID, eg_Whh, eg_Whh + (size_t)G4 * HID };
  const float* bia_s[4] = { ef_b, ef_b + G4, eg_b, eg_b + G4 };
  for (int i = 0; i < 4; ++i)
    pack_w_kernel<<<NB(1024 * G4), 256, 0, stream>>>(Wih_s[i], 512, Whh_s[i], 512, 512, 1024,
                                                     Bp_enc[i], 1024 * G4);
  pack_w_kernel<<<NB(1536 * G4), 256, 0, stream>>>(dec_Wih, 1024, dec_Whh, 512, 1024, 1536,
                                                   Bp_dec, 1536 * G4);
  pack_w_kernel<<<NB(1024 * 512), 256, 0, stream>>>(gi_W, 1024, gi_W, 1024, 1024, 1024,
                                                    Bp_gini, 1024 * 512);
  pack_w_kernel<<<NB(1024 * 512), 256, 0, stream>>>(asrc_W, 1024, asrc_W, 1024, 1024, 1024,
                                                    Bp_asrc, 1024 * 512);
  pack_w_kernel<<<NB(1536 * 512), 256, 0, stream>>>(avec_W, 1536, avec_W, 1536, 1536, 1536,
                                                    Bp_avec, 1536 * 512);

  // 3) four BiLSTM chains fused per step: 0=f-fwd 1=f-bwd 2=g-fwd 3=g-bwd
  const bf16_t* xsrc[4] = { x_bf, x_bf, y_bf, y_bf };
  for (int stp = 0; stp < SLEN; ++stp) {
    Enc4Args a;
    for (int ch = 0; ch < 4; ++ch) {
      int t = (ch & 1) ? (SLEN - 1 - stp) : stp;
      a.xseg[ch] = xsrc[ch] + (size_t)t * BATCH * DIM;
      a.hin[ch]  = hbf[ch][stp & 1];
      a.hout[ch] = hbf[ch][(stp + 1) & 1];
      a.c[ch]    = c_st[ch];
      a.h[ch]    = h_st[ch];
      a.oy[ch]   = (ch >= 2) ? outy_bf + (size_t)t * BATCH * 1024 + ((ch == 3) ? 512 : 0)
                             : nullptr;
      a.Bp[ch]   = Bp_enc[ch];
      a.bias[ch] = bia_s[ch];
    }
    enc4_kernel<<<dim3(32, 4), 256, 0, stream>>>(a);
  }

  // 4) decoder init: cell = [c_y_f;c_y_b]@g_init_W.T + b ; state = tanh(cell)
  f2bf_kernel<<<NB(32768), 256, 0, stream>>>(c_st[2], cyf_bf, 32768);
  f2bf_kernel<<<NB(32768), 256, 0, stream>>>(c_st[3], cyb_bf, 32768);
  gemm64_kernel<<<16, 256, 0, stream>>>(cyf_bf, 512, cyb_bf, 512, nullptr, 0, 2,
                                        Bp_gini, gi_b, dec_c, dec_h, dhbf[0], 1,
                                        nullptr, nullptr);

  // 5) src_att precompute: (16384x1024)@(1024x512) -> (B,S,H)
  srcatt_kernel<<<dim3(32, 128), 256, 0, stream>>>(outy_bf, Bp_asrc, src_att);

  // 6) decoder steps (cell -> attention -> att_vec GEMM with fused loss partial)
  for (int t = 0; t < SLEN; ++t) {
    cell_kernel<<<32, 256, 0, stream>>>(emb_bf + (size_t)t * BATCH * DIM,
                                        att_bf, dhbf[t & 1],
                                        Bp_dec, dec_b, dec_c, dec_h, dhbf[(t + 1) & 1]);
    attn_kernel<<<64, 256, 0, stream>>>(src_att, dec_h, outy_bf, lens, ctx_bf);
    gemm64_kernel<<<16, 256, 0, stream>>>(dhbf[(t + 1) & 1], 512, ctx_bf, 1024,
                                          ctx_bf + 512, 1024, 3,
                                          Bp_avec, nullptr, nullptr, att_f, att_bf, 1,
                                          x + (size_t)t * BATCH * DIM, ldp + t * 16);
  }

  // 7) losses
  losssq_kernel<<<32, 256, 0, stream>>>(h_st[0], h_st[2], lb);        // fwd final hiddens
  losssq_kernel<<<32, 256, 0, stream>>>(h_st[1], h_st[3], lb + 32);   // bwd final hiddens
  final_kernel<<<1, 256, 0, stream>>>(lb, ldp, out);
}
